// MHADecodeKernel_6691559047307
// MI455X (gfx1250) — matrix operand order repaired and checked by reading, not compiled, not run on hardware
//
#include <hip/hip_runtime.h>
#include <hip/hip_bf16.h>

typedef __attribute__((ext_vector_type(16))) __bf16       v16bf;
typedef __attribute__((ext_vector_type(8)))  float        v8f;
typedef __attribute__((ext_vector_type(4)))  float        f32x4;
typedef __attribute__((ext_vector_type(2)))  unsigned int u32x2;

constexpr int  B = 4, SQ = 128, SKV = 8192, H = 32, D = 128;
constexpr int  HD = H * D;                       // 4096
constexpr long BSQH  = (long)B * SQ * H;         // 16384
constexpr long BSQHD = (long)B * SQ * H * D;     // 2097152
constexpr int  NSPLIT_MAX = 4;
// softmax scale folded with log2(e): exp(x*s) == exp2(x*s*log2e)
constexpr float QK_SCALE = 0.08838834764831845f * 1.44269504088896340736f;

__device__ __forceinline__ unsigned short f2bf(float f) {
  unsigned int u = __builtin_bit_cast(unsigned int, f);
  u += 0x7FFFu + ((u >> 16) & 1u);               // round-to-nearest-even
  return (unsigned short)(u >> 16);
}

// Load a 16-element bf16 fragment as two 16B LDS reads (halves at +0 and +off2 elems).
__device__ __forceinline__ v16bf load_frag(const unsigned short* p, int off2) {
  union { v16bf v; f32x4 h[2]; } u;
  u.h[0] = *(const f32x4*)p;
  u.h[1] = *(const f32x4*)(p + off2);
  return u.v;
}

__device__ __forceinline__ v8f wmma_bf16(v16bf a, v16bf b, v8f c) {
  return __builtin_amdgcn_wmma_f32_16x16x32_bf16(false, a, false, b, (short)0, c,
                                                 false, false);
}

// ---- DPP16 butterfly reduction within each 16-lane row (no LDS traffic) ----
// 0xB1 = quad_perm[1,0,3,2] (xor1), 0x4E = quad_perm[2,3,0,1] (xor2),
// 0x141 = row_half_mirror (covers the other quad in each 8),
// 0x140 = row_mirror (covers the other 8 in each 16).
template <int CTRL>
__device__ __forceinline__ float dppf(float x) {
  int v = __builtin_amdgcn_update_dpp(0, __builtin_bit_cast(int, x),
                                      CTRL, 0xf, 0xf, true);
  return __builtin_bit_cast(float, v);
}
__device__ __forceinline__ float row_max16(float x) {
  x = fmaxf(x, dppf<0xB1>(x));
  x = fmaxf(x, dppf<0x4E>(x));
  x = fmaxf(x, dppf<0x141>(x));
  x = fmaxf(x, dppf<0x140>(x));
  return x;
}
__device__ __forceinline__ float row_sum16(float x) {
  x += dppf<0xB1>(x);
  x += dppf<0x4E>(x);
  x += dppf<0x141>(x);
  x += dppf<0x140>(x);
  return x;
}

// One workgroup: (b, h, kv-split). 8 waves x 16 q rows = SQ=128.
__global__ __launch_bounds__(256)
void fa_fwd(const float* __restrict__ qg, const float* __restrict__ kg,
            const float* __restrict__ vg, float* __restrict__ out,
            float* __restrict__ part, float* __restrict__ mbuf,
            float* __restrict__ lbuf, int nsplit, int kv_per_split) {
  __shared__ __align__(16) unsigned char smem[34816];
  unsigned short* Qs = (unsigned short*)smem;            // 128 x 136 (staging, reused)
  unsigned short* Kl = (unsigned short*)smem;            // 32 x 136  (row-major)
  unsigned short* Vt = (unsigned short*)(smem + 8704);   // 128 x 40  (d-major / transposed)
  const int tid  = threadIdx.x;
  const int lane = tid & 31;
  const int wave = tid >> 5;
  unsigned short* Pw = (unsigned short*)(smem + 18944) + wave * 640; // 16 x 40 per wave

  int id = blockIdx.x;
  const int split = id % nsplit;  id /= nsplit;
  const int h = id % H;
  const int b = id / H;
  const int kv0 = split * kv_per_split;
  const int kv1 = kv0 + kv_per_split;

  // ---- stage Q tile (128x128) as bf16, padded row stride 136 ----
  #pragma unroll
  for (int i = 0; i < 16; ++i) {
    int row = i * 8 + wave;
    int d   = lane * 4;
    f32x4 qv = *(const f32x4*)(qg + (((long)b * SQ + row) * H + h) * D + d);
    u32x2 pk;
    pk.x = (unsigned int)f2bf(qv.x) | ((unsigned int)f2bf(qv.y) << 16);
    pk.y = (unsigned int)f2bf(qv.z) | ((unsigned int)f2bf(qv.w) << 16);
    *(u32x2*)(Qs + row * 136 + d) = pk;
  }
  __syncthreads();

  const int halfsel = (lane & 16) ? 1 : 0;
  const int akb = halfsel ? 16 : 0;   // A-fragment K base (contiguous layout)
  const int bkb = halfsel ? 16 : 0;   // B-fragment K base (contiguous layout)
  const int n16 = lane & 15;

  // Q A-fragments: 16x128 per wave -> 4 x (16x32 bf16)
  v16bf Af[4];
  {
    const unsigned short* qr = Qs + (wave * 16 + n16) * 136 + akb;
    #pragma unroll
    for (int kk = 0; kk < 4; ++kk) Af[kk] = load_frag(qr + kk * 32, 8);
  }
  __syncthreads();   // Q staging area about to be reused for K/V

  v8f Of[8];
  float rm[8], rl[8];
  #pragma unroll
  for (int r = 0; r < 8; ++r) { rm[r] = -1e30f; rl[r] = 0.0f; }
  #pragma unroll
  for (int dc = 0; dc < 8; ++dc) Of[dc] = (v8f)0.0f;

  for (int s0 = kv0; s0 < kv1; s0 += 32) {
    // ---- stage K (row-major bf16) and V (transposed bf16) for 32 KV rows ----
    #pragma unroll
    for (int i = 0; i < 4; ++i) {
      int kvr = i * 8 + wave;
      int d   = lane * 4;
      long gro = (((long)b * SKV + (s0 + kvr)) * H + h) * D + d;
      f32x4 kf = *(const f32x4*)(kg + gro);
      f32x4 vf = *(const f32x4*)(vg + gro);
      __builtin_prefetch(kg + gro + 32L * HD, 0, 1);  // next tile -> global_prefetch_b8
      __builtin_prefetch(vg + gro + 32L * HD, 0, 1);
      u32x2 pk;
      pk.x = (unsigned int)f2bf(kf.x) | ((unsigned int)f2bf(kf.y) << 16);
      pk.y = (unsigned int)f2bf(kf.z) | ((unsigned int)f2bf(kf.w) << 16);
      *(u32x2*)(Kl + kvr * 136 + d) = pk;
      Vt[(d + 0) * 40 + kvr] = f2bf(vf.x);
      Vt[(d + 1) * 40 + kvr] = f2bf(vf.y);
      Vt[(d + 2) * 40 + kvr] = f2bf(vf.z);
      Vt[(d + 3) * 40 + kvr] = f2bf(vf.w);
    }
    __syncthreads();

    // ---- S = Q * K^T : two 16x16 chunks over K-dim 128 (4 x k32) ----
    v8f S0 = (v8f)0.0f, S1 = (v8f)0.0f;
    {
      const unsigned short* kr0 = Kl + n16 * 136 + bkb;
      const unsigned short* kr1 = Kl + (16 + n16) * 136 + bkb;
      #pragma unroll
      for (int kk = 0; kk < 4; ++kk) {
        S0 = wmma_bf16(Af[kk], load_frag(kr0 + kk * 32, 8), S0);
        S1 = wmma_bf16(Af[kk], load_frag(kr1 + kk * 32, 8), S1);
      }
    }

    // ---- online softmax (log2 domain); DPP butterfly over 16-lane halves ----
    float p0[8], p1[8], al[8];
    #pragma unroll
    for (int r = 0; r < 8; ++r) {
      float a = S0[r] * QK_SCALE;
      float c = S1[r] * QK_SCALE;
      float mx = row_max16(fmaxf(a, c));
      float mnew = fmaxf(rm[r], mx);
      al[r] = __builtin_amdgcn_exp2f(rm[r] - mnew);
      float pa = __builtin_amdgcn_exp2f(a - mnew);
      float pc = __builtin_amdgcn_exp2f(c - mnew);
      float sum = row_sum16(pa + pc);
      rl[r] = rl[r] * al[r] + sum;
      rm[r] = mnew;
      p0[r] = pa; p1[r] = pc;
    }
    #pragma unroll
    for (int dc = 0; dc < 8; ++dc)
      #pragma unroll
      for (int r = 0; r < 8; ++r) Of[dc][r] *= al[r];

    // ---- P: C-layout -> A-layout via per-wave LDS bounce ----
    {
      int prow = halfsel ? 8 : 0;
      #pragma unroll
      for (int r = 0; r < 8; ++r) {
        Pw[(prow + r) * 40 + n16]      = f2bf(p0[r]);
        Pw[(prow + r) * 40 + 16 + n16] = f2bf(p1[r]);
      }
    }
    asm volatile("" ::: "memory");  // wave-private LDS; HW keeps DS ops in order per wave
    v16bf Pf = load_frag(Pw + n16 * 40 + akb, 8);

    // ---- O += P * V : 8 d-chunks of 16 cols ----
    #pragma unroll
    for (int dc = 0; dc < 8; ++dc) {
      v16bf Vb = load_frag(Vt + (dc * 16 + n16) * 40 + bkb, 8);
      Of[dc] = wmma_bf16(Pf, Vb, Of[dc]);
    }
    __syncthreads();
  }

  // ---- epilogue ----
  const int rowoff = halfsel ? 8 : 0;
  if (nsplit == 1) {
    #pragma unroll
    for (int r = 0; r < 8; ++r) {
      int sq = wave * 16 + rowoff + r;
      float inv = 1.0f / rl[r];
      float* ob = out + (((long)b * SQ + sq) * H + h) * D;
      #pragma unroll
      for (int dc = 0; dc < 8; ++dc)
        ob[dc * 16 + n16] = Of[dc][r] * inv;
    }
  } else {
    #pragma unroll
    for (int r = 0; r < 8; ++r) {
      int sq  = wave * 16 + rowoff + r;
      long rid = ((long)b * SQ + sq) * H + h;
      float* ob = part + (long)split * BSQHD + rid * D;
      #pragma unroll
      for (int dc = 0; dc < 8; ++dc)
        ob[dc * 16 + n16] = Of[dc][r];
      if (n16 == 0) {
        mbuf[(long)split * BSQH + rid] = rm[r];
        lbuf[(long)split * BSQH + rid] = rl[r];
      }
    }
  }
}

// LSE combine across splits: out = sum_s w_s*O_s / sum_s w_s*l_s, w_s = 2^(m_s - m*)
__global__ __launch_bounds__(128)
void fa_combine(const float* __restrict__ part, const float* __restrict__ mbuf,
                const float* __restrict__ lbuf, float* __restrict__ out, int nsplit) {
  long rid = blockIdx.x;         // (b*SQ + sq)*H + h
  int  d   = threadIdx.x;        // 0..127
  float mmax = -1e30f;
  for (int s = 0; s < nsplit; ++s)
    mmax = fmaxf(mmax, mbuf[s * BSQH + rid]);
  float denom = 0.0f, acc = 0.0f;
  for (int s = 0; s < nsplit; ++s) {
    float w = exp2f(mbuf[s * BSQH + rid] - mmax);
    denom += w * lbuf[s * BSQH + rid];
    acc   += w * part[s * BSQHD + rid * D + d];
  }
  out[rid * D + d] = acc / denom;
}

extern "C" void kernel_launch(void* const* d_in, const int* in_sizes, int n_in,
                              void* d_out, int out_size, void* d_ws, size_t ws_size,
                              hipStream_t stream) {
  (void)in_sizes; (void)n_in; (void)out_size;
  const float* q = (const float*)d_in[0];
  const float* k = (const float*)d_in[1];
  const float* v = (const float*)d_in[2];
  float* out = (float*)d_out;

  size_t need = ((size_t)NSPLIT_MAX * BSQHD + 2ul * NSPLIT_MAX * BSQH) * sizeof(float);
  int nsplit = (d_ws != nullptr && ws_size >= need) ? NSPLIT_MAX : 1;

  float* part = (float*)d_ws;
  float* mb   = part + (long)nsplit * BSQHD;
  float* lb   = mb + (long)nsplit * BSQH;
  int kvps = SKV / nsplit;

  fa_fwd<<<dim3((unsigned)(B * H * nsplit)), dim3(256), 0, stream>>>(
      q, k, v, out, part, mb, lb, nsplit, kvps);
  if (nsplit > 1)
    fa_combine<<<dim3((unsigned)BSQH), dim3(128), 0, stream>>>(part, mb, lb, out, nsplit);
}
